// HetNet_8151847927966
// MI455X (gfx1250) — compile-verified, matrix operand order checked by reading, not run on hardware
//
#include <hip/hip_runtime.h>

typedef __attribute__((ext_vector_type(16))) __bf16 v16bf;
typedef __attribute__((ext_vector_type(8)))  __bf16 v8bf;
typedef __attribute__((ext_vector_type(8)))  float  v8f;

#define N_NODES 50000
#define DEG     8
#define E_EDGES (N_NODES*DEG)
#define D       128
#define S       46
#define TDIM    82
#define F       256
#define H       128
#define NOUT    2
#define KXH     256                 // [x | h] concat depth for LSTM GEMM
#define GATES   512
#define KFC     384                 // [x | emb] depth for fc1
#define XHP     (KXH + 8)           // padded LDS row stride (bf16) -> 528B, no bank aliasing
#define WSZ     (32*8*32*16)        // packed LSTM weight elements (131072 bf16 = 256KB)
#define FRAGE   (32*16)             // elements per packed B fragment (512 bf16 = 1KB)
#define LSTM_GRID 625
#define LSTM_GPW  5                 // groups per workgroup: 625*5*16 = 50000 rows
#define LSTM_SMEM (WSZ*2 + 16*XHP*2)

__device__ __forceinline__ float fsigmoid(float x){ return 1.0f/(1.0f+__expf(-x)); }
__device__ __forceinline__ float ftanh(float x){ float e = __expf(2.0f*x); return (e-1.0f)/(e+1.0f); }

// ---------------------------------------------------------------------------
// Stage 1: per-edge scores -> bf16, scores[e][j] = h_e[j]*rel[rels[e]][j]*t_e[j]
// ---------------------------------------------------------------------------
__global__ void edge_scores_kernel(const float* __restrict__ weeks, const float* __restrict__ days,
    const float* __restrict__ ent,  const float* __restrict__ rel,
    const float* __restrict__ wf,   const float* __restrict__ wp,  const float* __restrict__ wa,
    const float* __restrict__ df,   const float* __restrict__ dp,  const float* __restrict__ da,
    const int* __restrict__ heads,  const int* __restrict__ tails, const int* __restrict__ rels,
    __bf16* __restrict__ scores)
{
  int tid = threadIdx.x;
  int e   = blockIdx.x*2 + (tid >> 7);
  int j   = tid & 127;
  if (e >= E_EDGES) return;
  int h = heads[e], t = tails[e], r = rels[e];
  float wk = weeks[e], dy = days[e];
  float hv, tv;
  if (j < S) {
    hv = ent[h*S + j];
    tv = ent[t*S + j];
  } else {
    int jj = j - S;
    hv = wa[h*TDIM+jj]*__sinf(wf[h*TDIM+jj]*wk + wp[h*TDIM+jj])
       + da[h*TDIM+jj]*__sinf(df[h*TDIM+jj]*dy + dp[h*TDIM+jj]);
    tv = wa[t*TDIM+jj]*__sinf(wf[t*TDIM+jj]*wk + wp[t*TDIM+jj])
       + da[t*TDIM+jj]*__sinf(df[t*TDIM+jj]*dy + dp[t*TDIM+jj]);
  }
  float sc = hv * rel[r*D + j] * tv;
  scores[(size_t)e*D + j] = (__bf16)sc;
}

// ---------------------------------------------------------------------------
// Weight packing into WMMA B-fragment order (bf16):
// fragment f = (ct, kk); lane L: n = ct*16 + (L&15), kh = L>>4
// per-lane 16 bf16 = k-locals {kh*8+0..7, 16+kh*8+0..7}  ->  one 32B load/lane
// ---------------------------------------------------------------------------
__global__ void pack_lstm_w(const float* __restrict__ Wih, const float* __restrict__ Whh,
                            __bf16* __restrict__ out)
{
  int idx = blockIdx.x*256 + threadIdx.x;       // 32ct * 8kk * 32L * 16s = 131072
  int s  = idx & 15;
  int L  = (idx >> 4) & 31;
  int kk = (idx >> 9) & 7;
  int ct = idx >> 12;
  int n  = ct*16 + (L & 15);
  int kh = L >> 4;
  int kloc = (s < 8) ? (kh*8 + s) : (16 + kh*8 + (s - 8));
  int k = kk*32 + kloc;                          // 0..255 over [x|h]
  float v = (k < D) ? Wih[n*D + k] : Whh[n*D + (k - D)];
  out[idx] = (__bf16)v;
}

__global__ void pack_fc1_w(const float* __restrict__ fc1w, __bf16* __restrict__ out)
{
  int idx = blockIdx.x*256 + threadIdx.x;       // 8ct * 12kk * 32L * 16s = 49152
  int s  = idx & 15;
  int L  = (idx >> 4) & 31;
  int v  = idx >> 9;
  int kk = v % 12;
  int ct = v / 12;
  int n  = ct*16 + (L & 15);
  int kh = L >> 4;
  int kloc = (s < 8) ? (kh*8 + s) : (16 + kh*8 + (s - 8));
  int k = kk*32 + kloc;                          // 0..383
  out[idx] = (__bf16)fc1w[n*KFC + k];
}

__global__ void zero_f32(float* __restrict__ p, int n)
{
  int i = blockIdx.x*256 + threadIdx.x;
  if (i < n) p[i] = 0.0f;
}

// ---------------------------------------------------------------------------
// Stage 2: LSTM, weights LDS-resident (256KB), gate-aligned column tiles.
// 256 threads = 8 waves. Wave w owns tiles {w, w+8, w+16, w+24} == gates i/f/g/o
// for columns j = w*16 + lane%16  ->  nonlinearity fully in registers,
// c state in 8 VGPRs/lane, only h round-trips through LDS (it is next A).
// Accumulators start at 0; gate biases added scalar-wise post-GEMM (no long
// bias live range -> no spills). GEMM loop streams A per-kk to keep pressure low.
// ---------------------------------------------------------------------------
__global__ void lstm_kernel(const __bf16* __restrict__ scores, const int* __restrict__ seq,
                            const int* __restrict__ node_map, const __bf16* __restrict__ Wpack,
                            const float* __restrict__ bih, const float* __restrict__ bhh,
                            float* __restrict__ emb, int ngroups)
{
  extern __shared__ __align__(16) char smem[];
  __bf16* wlds = (__bf16*)smem;                       // WSZ bf16 = 256KB, same layout as Wpack
  __bf16* xh   = (__bf16*)(smem + (size_t)WSZ*2);     // [16][XHP] padded staging

  const int tid  = threadIdx.x;
  const int lane = tid & 31;
  const int wave = tid >> 5;
  const int li   = lane & 15;
  const int kh   = lane >> 4;
  const int srow = tid >> 4;
  const int sseg = tid & 15;

  // cooperative one-time fill of LDS weights (coalesced 16B chunks)
  for (int i = tid; i < WSZ/8; i += 256)
    *(v8bf*)(wlds + (size_t)i*8) = *(const v8bf*)(Wpack + (size_t)i*8);

  // per-lane gate biases for column jc (gates at +0/+128/+256/+384)
  const int jc = wave*16 + li;
  const float bI = bih[jc      ] + bhh[jc      ];
  const float bF = bih[jc + 128] + bhh[jc + 128];
  const float bG = bih[jc + 256] + bhh[jc + 256];
  const float bO = bih[jc + 384] + bhh[jc + 384];

  // per-gate LDS weight base pointers: fragment (ct,kk) at (ct*8+kk)*FRAGE + lane*16
  const __bf16* wI = wlds + (size_t)((wave      )*8)*FRAGE + (size_t)lane*16;
  const __bf16* wF = wlds + (size_t)((wave +  8 )*8)*FRAGE + (size_t)lane*16;
  const __bf16* wG = wlds + (size_t)((wave + 16 )*8)*FRAGE + (size_t)lane*16;
  const __bf16* wO = wlds + (size_t)((wave + 24 )*8)*FRAGE + (size_t)lane*16;
  const __bf16* arow = xh + li*XHP + kh*8;            // this lane's A-row base

  __syncthreads();

  for (int grp = 0; grp < ngroups; ++grp) {
    const int base = (blockIdx.x * ngroups + grp) * 16;

    // output node ids for this group's rows owned by this lane (row = 8*kh + r)
    int nodes[8];
    #pragma unroll
    for (int r = 0; r < 8; ++r)
      nodes[r] = node_map[seq[(base + 8*kh + r)*DEG]];

    float creg[8];
    #pragma unroll
    for (int r = 0; r < 8; ++r) creg[r] = 0.0f;

    for (int t = 0; t < DEG; ++t) {
      // ---- stage x_t (gathered rows of scores); zero h-region at t==0 ----
      {
        int e = seq[(base + srow)*DEG + t];
        *(v8bf*)&xh[srow*XHP + sseg*8] = *(const v8bf*)(scores + (size_t)e*D + sseg*8);
        if (t == 0) {
          v8bf z;
          #pragma unroll
          for (int i = 0; i < 8; ++i) z[i] = (__bf16)0.0f;
          *(v8bf*)&xh[srow*XHP + D + sseg*8] = z;
        }
      }
      __syncthreads();                 // x|h staged, prior-step h-writes visible

      // ---- GEMM: stream A per K-chunk, 4 independent gate chains ----
      v8f aI = {}, aF = {}, aG = {}, aO = {};
      #pragma unroll
      for (int kk = 0; kk < 8; ++kk) {
        v8bf lo = *(const v8bf*)(arow + kk*32);
        v8bf hi = *(const v8bf*)(arow + kk*32 + 16);
        v16bf a;
        #pragma unroll
        for (int i = 0; i < 8; ++i) { a[i] = lo[i]; a[8+i] = hi[i]; }
        v16bf bi = *(const v16bf*)(wI + kk*FRAGE);
        v16bf bf = *(const v16bf*)(wF + kk*FRAGE);
        v16bf bg = *(const v16bf*)(wG + kk*FRAGE);
        v16bf bo = *(const v16bf*)(wO + kk*FRAGE);
        aI = __builtin_amdgcn_wmma_f32_16x16x32_bf16(false, a, false, bi, (short)0, aI, false, false);
        aF = __builtin_amdgcn_wmma_f32_16x16x32_bf16(false, a, false, bf, (short)0, aF, false, false);
        aG = __builtin_amdgcn_wmma_f32_16x16x32_bf16(false, a, false, bg, (short)0, aG, false, false);
        aO = __builtin_amdgcn_wmma_f32_16x16x32_bf16(false, a, false, bo, (short)0, aO, false, false);
      }
      __syncthreads();                 // all xh reads for this step complete

      // ---- LSTM nonlinearity fully in registers (biases added here) ----
      if (t == DEG-1) {
        #pragma unroll
        for (int r = 0; r < 8; ++r) {
          float c  = fsigmoid(aF[r]+bF)*creg[r] + fsigmoid(aI[r]+bI)*ftanh(aG[r]+bG);
          float hv = fsigmoid(aO[r]+bO)*ftanh(c);
          emb[(size_t)nodes[r]*D + jc] = hv;        // final h -> emb (no LDS write-back)
        }
      } else {
        #pragma unroll
        for (int r = 0; r < 8; ++r) {
          float c  = fsigmoid(aF[r]+bF)*creg[r] + fsigmoid(aI[r]+bI)*ftanh(aG[r]+bG);
          creg[r] = c;
          float hv = fsigmoid(aO[r]+bO)*ftanh(c);
          xh[(8*kh + r)*XHP + D + jc] = (__bf16)hv; // h -> next step's A
        }
      }
      // no trailing barrier: next stage writes x/h regions only, and all xh
      // reads for this step completed before the barrier above.
    }
  }
}

// ---------------------------------------------------------------------------
// Stage 3: h1 = [x|emb] @ fc1_w^T + b  (WMMA), LayerNorm, ReLU, out = h1 @ out_w^T + b
// ---------------------------------------------------------------------------
__global__ void mlp_kernel(const float* __restrict__ x, const float* __restrict__ emb,
                           const __bf16* __restrict__ Fpack,
                           const float* __restrict__ fc1_b, const float* __restrict__ ln_g,
                           const float* __restrict__ ln_b,  const float* __restrict__ out_w,
                           const float* __restrict__ out_b, float* __restrict__ out)
{
  __shared__ __align__(16) __bf16 xc[16][KFC + 8];
  __shared__ float h1[16][H];

  const int tid  = threadIdx.x;
  const int lane = tid & 31;
  const int wave = tid >> 5;
  const int base = blockIdx.x * 16;

  for (int idx = tid; idx < 16*KFC; idx += 256) {
    int row = idx / KFC;
    int j   = idx % KFC;
    float v = (j < F) ? x[(size_t)(base+row)*F + j] : emb[(size_t)(base+row)*D + (j - F)];
    xc[row][j] = (__bf16)v;
  }
  __syncthreads();

  {
    const int m    = lane & 15;
    const int kh   = lane >> 4;
    const int ct   = wave;               // 8 tiles of 16 -> 128 outputs
    const int ncol = ct*16 + (lane & 15);
    v8f acc = {};
    const v16bf* wptr = (const v16bf*)Fpack + (size_t)(ct*12)*32 + lane;
    #pragma unroll
    for (int kk = 0; kk < 12; ++kk) {
      const __bf16* p = &xc[m][kk*32 + kh*8];
      v8bf lo = *(const v8bf*)p;
      v8bf hi = *(const v8bf*)(p + 16);
      v16bf a;
      #pragma unroll
      for (int i = 0; i < 8; ++i) { a[i] = lo[i]; a[8+i] = hi[i]; }
      acc = __builtin_amdgcn_wmma_f32_16x16x32_bf16(false, a, false, wptr[kk*32],
                                                    (short)0, acc, false, false);
    }
    const float bias = fc1_b[ncol];
    const int mb = 8*(lane >> 4);
    #pragma unroll
    for (int r = 0; r < 8; ++r) h1[mb + r][ncol] = acc[r] + bias;
  }
  __syncthreads();

  // LayerNorm + ReLU + tiny output GEMV; wave w handles rows 2w (lanes 0-15) and 2w+1 (16-31)
  {
    const int li = lane & 15;
    const int m  = wave*2 + (lane >> 4);
    float vals[8];
    float s = 0.0f;
    #pragma unroll
    for (int q = 0; q < 8; ++q) { vals[q] = h1[m][li*8 + q]; s += vals[q]; }
    #pragma unroll
    for (int msk = 8; msk >= 1; msk >>= 1) s += __shfl_xor(s, msk, 16);
    float mean = s * (1.0f/128.0f);
    float vs = 0.0f;
    #pragma unroll
    for (int q = 0; q < 8; ++q) { float d2 = vals[q] - mean; vs += d2*d2; }
    #pragma unroll
    for (int msk = 8; msk >= 1; msk >>= 1) vs += __shfl_xor(vs, msk, 16);
    float inv = rsqrtf(vs * (1.0f/128.0f) + 1e-5f);
    float d0 = 0.0f, d1 = 0.0f;
    #pragma unroll
    for (int q = 0; q < 8; ++q) {
      int j = li*8 + q;
      float hv = (vals[q] - mean)*inv*ln_g[j] + ln_b[j];
      hv = fmaxf(hv, 0.0f);
      d0 += hv * out_w[j];
      d1 += hv * out_w[H + j];
    }
    #pragma unroll
    for (int msk = 8; msk >= 1; msk >>= 1) { d0 += __shfl_xor(d0, msk, 16); d1 += __shfl_xor(d1, msk, 16); }
    if (li == 0) {
      int node = base + m;
      out[(size_t)node*NOUT + 0] = d0 + out_b[0];
      out[(size_t)node*NOUT + 1] = d1 + out_b[1];
    }
  }
}

// ---------------------------------------------------------------------------
extern "C" void kernel_launch(void* const* d_in, const int* in_sizes, int n_in,
                              void* d_out, int out_size, void* d_ws, size_t ws_size,
                              hipStream_t stream)
{
  const float* x     = (const float*)d_in[0];
  const float* weeks = (const float*)d_in[1];
  const float* days  = (const float*)d_in[2];
  const float* ent   = (const float*)d_in[3];
  const float* rel   = (const float*)d_in[4];
  const float* wf    = (const float*)d_in[5];
  const float* wp    = (const float*)d_in[6];
  const float* wa    = (const float*)d_in[7];
  const float* df    = (const float*)d_in[8];
  const float* dp    = (const float*)d_in[9];
  const float* da    = (const float*)d_in[10];
  const float* Wih   = (const float*)d_in[11];
  const float* Whh   = (const float*)d_in[12];
  const float* bih   = (const float*)d_in[13];
  const float* bhh   = (const float*)d_in[14];
  const float* fc1w  = (const float*)d_in[15];
  const float* fc1b  = (const float*)d_in[16];
  const float* lng   = (const float*)d_in[17];
  const float* lnb   = (const float*)d_in[18];
  const float* outw  = (const float*)d_in[19];
  const float* outb  = (const float*)d_in[20];
  const int* heads   = (const int*)d_in[21];
  const int* rels    = (const int*)d_in[22];
  const int* tails   = (const int*)d_in[23];
  const int* src_idx = (const int*)d_in[24];
  const int* dst_idx = (const int*)d_in[25];
  float* out = (float*)d_out;

  char* ws = (char*)d_ws;
  size_t off = 0;
  __bf16* scores = (__bf16*)(ws + off); off += (size_t)E_EDGES * D * sizeof(__bf16);   // 102.4 MB
  __bf16* Wpack  = (__bf16*)(ws + off); off += (size_t)WSZ * sizeof(__bf16);           // 256 KB
  __bf16* Fpack  = (__bf16*)(ws + off); off += (size_t)8*12*32*16 * sizeof(__bf16);    // 96 KB
  off = (off + 255) & ~(size_t)255;
  float*  embp   = (float*)(ws + off);  off += (size_t)N_NODES * D * sizeof(float);    // 25.6 MB

  edge_scores_kernel<<<E_EDGES/2, 256, 0, stream>>>(weeks, days, ent, rel,
                                                    wf, wp, wa, df, dp, da,
                                                    heads, tails, rels, scores);
  pack_lstm_w<<<(WSZ)/256, 256, 0, stream>>>(Wih, Whh, Wpack);
  pack_fc1_w<<<(8*12*32*16)/256, 256, 0, stream>>>(fc1w, Fpack);
  zero_f32<<<(N_NODES*D + 255)/256, 256, 0, stream>>>(embp, N_NODES*D);

  // src pass first, dst pass second: stream order reproduces the .at[].set override
  lstm_kernel<<<LSTM_GRID, 256, LSTM_SMEM, stream>>>(scores, src_idx, heads, Wpack, bih, bhh, embp, LSTM_GPW);
  lstm_kernel<<<LSTM_GRID, 256, LSTM_SMEM, stream>>>(scores, dst_idx, tails, Wpack, bih, bhh, embp, LSTM_GPW);

  mlp_kernel<<<N_NODES/16, 256, 0, stream>>>(x, embp, Fpack, fc1b, lng, lnb, outw, outb, out);
}
